// PartialGConv_17592186044549
// MI455X (gfx1250) — compile-verified, matrix operand order checked by reading, not run on hardware
//
#include <hip/hip_runtime.h>
#include <hip/hip_bf16.h>

// ---------------------------------------------------------------------------
// PartialGConv for MI455X (gfx1250, wave32, WMMA).
// Stage 1: temporal partial-conv as 9 accumulating bf16 WMMA GEMMs (K=64 each),
//          M=192 out-channels, positions flattened over (t,v) so the temporal
//          shift is a flat +/- dt*25 index with zero padding at the edges.
// Stage 2: graph einsum y2 = sum_{k,v} out * A as a skinny bf16 WMMA GEMM
//          (rows=(t,c), K=(k,v) padded 75->96).
// Both use v_wmma_f32_16x16x32_bf16 with f32 accumulation.
// ---------------------------------------------------------------------------

typedef __attribute__((ext_vector_type(16))) __bf16 v16bf;
typedef __attribute__((ext_vector_type(8)))  __bf16 v8bf;
typedef __attribute__((ext_vector_type(8)))  float  v8f;

constexpr int NB   = 32;
constexpr int CIN  = 64;
constexpr int COUT = 64;
constexpr int TT   = 256;
constexpr int VV   = 25;
constexpr int KK   = 3;
constexpr int TKL  = 9;
constexpr int KC   = KK * COUT;      // 192
constexpr int PVn  = TT * VV;        // 6400 flat (t,v) positions per (n,ci)
constexpr float EPSF  = 1e-8f;
constexpr float SLIDE = (float)(CIN * TKL); // 576

// workspace layout (bytes)
constexpr size_t O_XM    = 0;                                   // bf16 xm  (N,Cin,T*V)
constexpr size_t SZ_XM   = (size_t)NB * CIN * PVn * 2;          // 26,214,400
constexpr size_t O_WT    = O_XM + SZ_XM;                        // bf16 wt  [dt][kc][ci]
constexpr size_t SZ_WT   = (size_t)TKL * KC * CIN * 2;          // 221,184
constexpr size_t O_MSUM  = O_WT + SZ_WT;                        // f32 msum (N,T*V)
constexpr size_t SZ_NP   = (size_t)NB * PVn * 4;                // 819,200
constexpr size_t O_RATIO = O_MSUM + SZ_NP;
constexpr size_t O_UPDC  = O_RATIO + SZ_NP;
constexpr size_t O_PCONV = O_UPDC + SZ_NP;                      // bf16 pconv (N,192,T*V)

constexpr long long Y_ELEMS = (long long)NB * COUT * TT * VV;   // 13,107,200
constexpr int A_ELEMS = KK * VV * VV;                           // 1875

__device__ inline v16bf pack16(v8bf lo, v8bf hi) {
    v16bf r;
#pragma unroll
    for (int i = 0; i < 8; ++i) { r[i] = lo[i]; r[i + 8] = hi[i]; }
    return r;
}
__device__ inline v8f zero8() {
    v8f z;
#pragma unroll
    for (int i = 0; i < 8; ++i) z[i] = 0.f;
    return z;
}

// ---------------------------- prep kernels ---------------------------------

__global__ __launch_bounds__(256) void k_prep_xm(const float* __restrict__ x,
                                                 const float* __restrict__ mask,
                                                 __bf16* __restrict__ xm,
                                                 long long total) {
    long long i = (long long)blockIdx.x * 256 + threadIdx.x;
    if (i < total) xm[i] = (__bf16)(x[i] * mask[i]);
}

__global__ __launch_bounds__(256) void k_msum(const float* __restrict__ mask,
                                              float* __restrict__ msum) {
    int i = blockIdx.x * 256 + threadIdx.x;
    if (i >= NB * PVn) return;
    int n = i / PVn, p = i % PVn;
    const float* mp = mask + (size_t)n * CIN * PVn + p;
    float s = 0.f;
    for (int ci = 0; ci < CIN; ++ci) s += mp[(size_t)ci * PVn];
    msum[i] = s;
}

__global__ __launch_bounds__(256) void k_updratio(const float* __restrict__ msum,
                                                  float* __restrict__ ratio,
                                                  float* __restrict__ updc) {
    int i = blockIdx.x * 256 + threadIdx.x;
    if (i >= NB * PVn) return;
    int n = i / PVn, p = i % PVn;
    float u = 0.f;
#pragma unroll
    for (int d = 0; d < TKL; ++d) {
        int q = p + (d - 4) * VV;            // temporal shift in flat (t,v)
        if (q >= 0 && q < PVn) u += msum[(size_t)n * PVn + q];
    }
    float uc = fminf(fmaxf(u, 0.f), 1.f);
    ratio[i] = SLIDE / (u + EPSF) * uc;
    updc[i]  = uc;
}

__global__ __launch_bounds__(256) void k_wtrans(const float* __restrict__ w,
                                                __bf16* __restrict__ wt) {
    int i = blockIdx.x * 256 + threadIdx.x;
    if (i >= TKL * KC * CIN) return;
    int dt = i / (KC * CIN), rem = i % (KC * CIN);
    int kc = rem / CIN, ci = rem % CIN;
    wt[i] = (__bf16)w[(size_t)kc * CIN * TKL + (size_t)ci * TKL + dt];
}

__global__ __launch_bounds__(256) void k_copyA(const float* __restrict__ A,
                                               float* __restrict__ outA) {
    int i = blockIdx.x * 256 + threadIdx.x;
    if (i < A_ELEMS) outA[i] = A[i];
}

// ------------------- stage 1: partial conv as WMMA GEMM --------------------
// Block: (position tile of 64, n). 8 waves; wave = 3 M-tiles (48 kc) x 2 N-tiles
// (32 positions). K loop: 9 temporal taps x 2 ci-chunks of 32.

__global__ __launch_bounds__(256) void k_pconv(const __bf16* __restrict__ xm,
                                               const __bf16* __restrict__ wt,
                                               const float*  __restrict__ ratio,
                                               const float*  __restrict__ updc,
                                               const float*  __restrict__ bias,
                                               __bf16* __restrict__ pout) {
    const int n  = blockIdx.y;
    const int pb = blockIdx.x * 64;

    __shared__ __bf16 lb[64 * 64];       // B staging: [p][ci], XOR-swizzled cols
    __shared__ float  lsr[64], lsu[64], lsb[KC];

    const int tid  = threadIdx.x;
    const int lane = tid & 31;
    const int wv   = tid >> 5;
    const int m16  = lane & 15;
    const int hf   = lane >> 4;          // selects K-chunk pair per ISA layout
    const int mbase = (wv >> 1) * 48;    // kc base for this wave
    const int nbase = (wv & 1) * 32;     // local position base

    if (tid < 64) {
        lsr[tid] = ratio[(size_t)n * PVn + pb + tid];
        lsu[tid] = updc [(size_t)n * PVn + pb + tid];
    }
    if (tid < KC) lsb[tid] = bias[tid];

    const __bf16* xmn = xm + (size_t)n * CIN * PVn;

    v8f acc[3][2];
#pragma unroll
    for (int i = 0; i < 3; ++i)
#pragma unroll
        for (int j = 0; j < 2; ++j) acc[i][j] = zero8();

    const int pl  = tid & 63;                       // local position this thread stages
    const int cig = tid >> 6;                       // ci group 0..3
    const int swS = ((pl ^ (pl >> 3)) & 7) << 3;    // bank swizzle for column pl

    for (int dt = 0; dt < TKL; ++dt) {
        const int  shift = (dt - 4) * VV;
        const int  gp    = pb + shift + pl;
        const bool ok    = (gp >= 0) && (gp < PVn);

        __syncthreads();
        // coalesced global read (consecutive lanes -> consecutive positions),
        // transposed + swizzled store into LDS
#pragma unroll
        for (int cc = 0; cc < 16; ++cc) {
            const int ci = cig * 16 + cc;
            __bf16 v = ok ? xmn[(size_t)ci * PVn + gp] : (__bf16)0.f;
            lb[pl * 64 + (ci ^ swS)] = v;
        }
        if (dt < TKL - 1) {                          // prefetch next temporal window
            int gq = gp + VV;
            if (gq < 0) gq = 0;
            if (gq >= PVn) gq = PVn - 1;
            __builtin_prefetch(xmn + (size_t)(cig * 16) * PVn + gq, 0, 1);
        }
        __syncthreads();

#pragma unroll
        for (int kt = 0; kt < 2; ++kt) {             // ci chunks of 32
            // A fragments (weights) straight from global: [dt][kc][ci], 16B aligned
            v16bf af[3];
#pragma unroll
            for (int i = 0; i < 3; ++i) {
                const __bf16* wr = wt
                    + ((size_t)(dt * KC + mbase + i * 16 + m16) << 6)
                    + kt * 32 + 8 * hf;
                af[i] = pack16(*(const v8bf*)wr, *(const v8bf*)(wr + 16));
            }
            // B fragments from LDS (ds_load_b128 pairs, swizzle-matched)
            v16bf bfr[2];
#pragma unroll
            for (int j = 0; j < 2; ++j) {
                const int p  = nbase + j * 16 + m16;
                const int sw = ((p ^ (p >> 3)) & 7) << 3;
                const int c0 = (kt * 32 + 8 * hf) ^ sw;
                const int c1 = (kt * 32 + 16 + 8 * hf) ^ sw;
                bfr[j] = pack16(*(const v8bf*)&lb[p * 64 + c0],
                                *(const v8bf*)&lb[p * 64 + c1]);
            }
#pragma unroll
            for (int i = 0; i < 3; ++i)
#pragma unroll
                for (int j = 0; j < 2; ++j)
                    acc[i][j] = __builtin_amdgcn_wmma_f32_16x16x32_bf16(
                        false, af[i], false, bfr[j], (short)0, acc[i][j],
                        false, false);
        }
    }

    // epilogue: out = (raw*ratio + bias) * upd_c, store bf16 for stage 2
    __bf16* outn = pout + (size_t)n * KC * PVn + pb;
#pragma unroll
    for (int j = 0; j < 2; ++j) {
        const int   p = nbase + j * 16 + m16;
        const float r = lsr[p], u = lsu[p];
#pragma unroll
        for (int i = 0; i < 3; ++i) {
#pragma unroll
            for (int e = 0; e < 8; ++e) {
                const int   kc  = mbase + i * 16 + e + 8 * hf;  // C/D VGPR layout
                const float val = (acc[i][j][e] * r + lsb[kc]) * u;
                outn[(size_t)kc * PVn + p] = (__bf16)val;
            }
        }
    }
}

// ------------------- stage 2: graph einsum as WMMA GEMM --------------------
// Block: (t pair, n). Rows = (t_local, c) -> 128; K = (k, v) padded 75 -> 96;
// cols = w padded 25 -> 32. 8 waves, each 1 M-tile x 2 N-tiles, K loop of 3.

__global__ __launch_bounds__(256) void k_einsum(const __bf16* __restrict__ pconv,
                                                const float*  __restrict__ Am,
                                                const float*  __restrict__ updc,
                                                float* __restrict__ yout,
                                                float* __restrict__ mout) {
    const int n  = blockIdx.y;
    const int t0 = blockIdx.x * 2;

    __shared__ __bf16 lv[128 * 96];   // A operand: [row=(tl,c)][k*32+v], zero-padded
    __shared__ __bf16 lA[32 * 96];    // B operand: [w][k*32+v], zero-padded
    __shared__ float  lm2[64];        // m2 per (tl, w)

    const int tid  = threadIdx.x;
    const int lane = tid & 31, wv = tid >> 5;
    const int m16  = lane & 15, hf = lane >> 4;

    for (int i = tid; i < 32 * 96;  i += 256) lA[i] = (__bf16)0.f;
    for (int i = tid; i < 128 * 96; i += 256) lv[i] = (__bf16)0.f;
    __syncthreads();

    for (int i = tid; i < KK * VV * VV; i += 256) {
        const int k = i / (VV * VV), rem = i % (VV * VV);
        const int v = rem / VV, w = rem % VV;
        lA[w * 96 + k * 32 + v] = (__bf16)Am[i];
    }
    for (int i = tid; i < 128 * KK * VV; i += 256) {
        const int v  = i % VV;
        const int r2 = i / VV;
        const int k  = r2 >> 7;
        const int row = r2 & 127;
        const int tl = row >> 6, c = row & 63;
        lv[row * 96 + k * 32 + v] =
            pconv[((size_t)n * KC + k * COUT + c) * PVn + (t0 + tl) * VV + v];
    }
    if (tid < 64) {   // m2[tl][w] = sum_v upd_c[t,v] * sum_k A[k,v,w]
        const int tl = tid >> 5, w = tid & 31;
        float s = 0.f;
        if (w < VV) {
#pragma unroll
            for (int v = 0; v < VV; ++v) {
                const float uv = updc[(size_t)n * PVn + (t0 + tl) * VV + v];
                const float a  = Am[v * VV + w] + Am[625 + v * VV + w]
                               + Am[1250 + v * VV + w];
                s += uv * a;
            }
        }
        lm2[tid] = s;
    }
    __syncthreads();

    v8f acc[2];
    acc[0] = zero8(); acc[1] = zero8();

#pragma unroll
    for (int kt = 0; kt < 3; ++kt) {
        const int row = wv * 16 + m16;
        const int r0  = kt * 32 + 8 * hf;
        v16bf af = pack16(*(const v8bf*)&lv[row * 96 + r0],
                          *(const v8bf*)&lv[row * 96 + r0 + 16]);
#pragma unroll
        for (int j = 0; j < 2; ++j) {
            const int w = j * 16 + m16;
            v16bf bfr = pack16(*(const v8bf*)&lA[w * 96 + r0],
                               *(const v8bf*)&lA[w * 96 + r0 + 16]);
            acc[j] = __builtin_amdgcn_wmma_f32_16x16x32_bf16(
                false, af, false, bfr, (short)0, acc[j], false, false);
        }
    }

#pragma unroll
    for (int j = 0; j < 2; ++j) {
        const int w = j * 16 + m16;
        if (w < VV) {
#pragma unroll
            for (int e = 0; e < 8; ++e) {
                const int row = wv * 16 + e + 8 * hf;
                const int tl = row >> 6, c = row & 63;
                const float m2v = lm2[tl * 32 + w];
                const float inv = (m2v != 0.f) ? 1.f / (m2v + EPSF) : 0.f;
                const size_t idx =
                    (((size_t)n * COUT + c) * TT + (t0 + tl)) * VV + w;
                yout[idx] = acc[j][e] * inv;
                mout[idx] = (m2v != 0.f) ? 1.f : 0.f;
            }
        }
    }
}

// ------------------------------- launcher ----------------------------------

extern "C" void kernel_launch(void* const* d_in, const int* in_sizes, int n_in,
                              void* d_out, int out_size, void* d_ws, size_t ws_size,
                              hipStream_t stream) {
    const float* x      = (const float*)d_in[0];
    const float* A      = (const float*)d_in[1];
    const float* mask   = (const float*)d_in[2];
    const float* weight = (const float*)d_in[3];
    const float* bias   = (const float*)d_in[4];

    char* ws = (char*)d_ws;
    __bf16* xm    = (__bf16*)(ws + O_XM);
    __bf16* wt    = (__bf16*)(ws + O_WT);
    float*  msum  = (float*) (ws + O_MSUM);
    float*  ratio = (float*) (ws + O_RATIO);
    float*  updc  = (float*) (ws + O_UPDC);
    __bf16* pconv = (__bf16*)(ws + O_PCONV);

    float* out  = (float*)d_out;
    float* yout = out;
    float* Aout = out + Y_ELEMS;
    float* mout = out + Y_ELEMS + A_ELEMS;

    const long long total_xm = (long long)NB * CIN * PVn;   // 13,107,200

    k_prep_xm <<<dim3((unsigned)((total_xm + 255) / 256)), dim3(256), 0, stream>>>(
        x, mask, xm, total_xm);
    k_msum    <<<dim3((NB * PVn + 255) / 256), dim3(256), 0, stream>>>(mask, msum);
    k_updratio<<<dim3((NB * PVn + 255) / 256), dim3(256), 0, stream>>>(msum, ratio, updc);
    k_wtrans  <<<dim3((TKL * KC * CIN + 255) / 256), dim3(256), 0, stream>>>(weight, wt);
    k_copyA   <<<dim3((A_ELEMS + 255) / 256), dim3(256), 0, stream>>>(A, Aout);

    k_pconv   <<<dim3(PVn / 64, NB), dim3(256), 0, stream>>>(
        xm, wt, ratio, updc, bias, pconv);

    k_einsum  <<<dim3(TT / 2, NB), dim3(256), 0, stream>>>(
        pconv, A, updc, yout, mout);
}
